// Decoder_72215580115112
// MI455X (gfx1250) — matrix operand order repaired and checked by reading, not compiled, not run on hardware
//
#include <hip/hip_runtime.h>

// ---------------------------------------------------------------------------
// CDNA5 (gfx1250) wave32 WMMA decoder: ASPP (1x1 convs) + bilinear upsample +
// 3x3 fuse conv + LIIF local-ensemble MLP. GEMMs via v_wmma_f32_16x16x32_f16,
// global->LDS staging via GLOBAL_LOAD_ASYNC_TO_LDS_B128 when available.
// ---------------------------------------------------------------------------

typedef _Float16 v16h  __attribute__((ext_vector_type(16)));
typedef float    v8f   __attribute__((ext_vector_type(8)));
typedef float    f32x4 __attribute__((ext_vector_type(4)));
typedef float    f32x2 __attribute__((ext_vector_type(2)));

union Frag16 { v16h h; f32x4 q[2]; };

#define WMMA_F16F32(A, B, C) \
  __builtin_amdgcn_wmma_f32_16x16x32_f16(false, (A), false, (B), (short)0, (C), false, false)

#if defined(__has_builtin)
#if __has_builtin(__builtin_amdgcn_global_load_async_to_lds_b128)
#define HAVE_ASYNC_LDS 1
#endif
#endif

#if defined(__has_builtin)
#if __has_builtin(__builtin_amdgcn_s_wait_asynccnt)
#define WAIT_ASYNC() __builtin_amdgcn_s_wait_asynccnt(0)
#elif defined(HAVE_ASYNC_LDS)
#define WAIT_ASYNC() asm volatile("s_wait_asynccnt 0" ::: "memory")
#else
#define WAIT_ASYNC()
#endif
#else
#define WAIT_ASYNC()
#endif

// 16-byte global -> LDS copy (async when the CDNA5 builtin is available).
// Builtin signature (from hipcc diagnostic): param0 = int __vector(4) __device__*,
// param1 = LDS pointer, params 2/3 = immediate offset / cpol.
__device__ __forceinline__ void copy16_g2l(const _Float16* g, _Float16* l) {
#ifdef HAVE_ASYNC_LDS
  typedef __attribute__((__vector_size__(4 * sizeof(int)))) int v4i_g;
  typedef __attribute__((address_space(1))) v4i_g* gp_t;
  typedef __attribute__((address_space(3))) v4i_g* lp_t;
  __builtin_amdgcn_global_load_async_to_lds_b128(
      (gp_t)(unsigned long long)(uintptr_t)g,
      (lp_t)(unsigned)(uintptr_t)l, 0, 0);
#else
  *(f32x4*)l = *(const f32x4*)g;
#endif
}

#define QN   262144        // 512*512 queries
#define FH   256           // feature map H=W
#define FC   96            // fused channels
#define MLP_U 128

// ---------------------------------------------------------------------------
// Weight convert+transpose: out[b][n][kpad] (f16) = in[b][k][n] (f32), zero pad.
// ---------------------------------------------------------------------------
__global__ void cvtT_kernel(const float* __restrict__ in, _Float16* __restrict__ out,
                            int K, int N, int Kpad, int batch) {
  int idx = blockIdx.x * blockDim.x + threadIdx.x;
  int total = batch * N * Kpad;
  if (idx >= total) return;
  int b = idx / (N * Kpad);
  int rem = idx - b * (N * Kpad);
  int n = rem / Kpad;
  int k = rem - n * Kpad;
  float v = (k < K) ? in[((size_t)b * K + k) * N + n] : 0.0f;
  out[idx] = (_Float16)v;
}

// ---------------------------------------------------------------------------
// Generic 1x1 conv + BN affine + ReLU as WMMA GEMM: [M x K] f32 @ [K x 32] -> f16
// wT layout: [32][K] f16 (N-major). K multiple of 32, M multiple of 64.
// ---------------------------------------------------------------------------
__global__ __launch_bounds__(128)
void conv1x1_wmma(const float* __restrict__ in, const _Float16* __restrict__ wT,
                  const float* __restrict__ scale, const float* __restrict__ bias,
                  _Float16* __restrict__ out, int K, int outStride, int outOff) {
  __shared__ __align__(16) _Float16 shA[64 * 256];

  const int tid  = threadIdx.x;
  const size_t gRowBase = (size_t)blockIdx.x * 64;
  const int kd2 = K >> 1;

  // stage 64 rows of f32 input as f16 into LDS
  for (int i = tid; i < 64 * kd2; i += 128) {
    int row = i / kd2;
    int c2  = i - row * kd2;
    f32x2 v = *(const f32x2*)(in + (gRowBase + row) * (size_t)K + 2 * c2);
    union { _Float16 h[2]; unsigned u; } P;
    P.h[0] = (_Float16)v.x; P.h[1] = (_Float16)v.y;
    *(unsigned*)&shA[row * K + 2 * c2] = P.u;
  }
  __syncthreads();

  const int wave = tid >> 5, lane = tid & 31;
  const int ln = lane & 15, hi = lane >> 4;
  const int rowBase = wave * 16;

  v8f acc[2] = {};
  const int nsteps = K >> 5;
  for (int s = 0; s < nsteps; ++s) {
    Frag16 a;
    const _Float16* ap = &shA[(rowBase + ln) * K + 32 * s + hi * 8];
    a.q[0] = *(const f32x4*)(ap);
    a.q[1] = *(const f32x4*)(ap + 16);
#pragma unroll
    for (int j = 0; j < 2; ++j) {
      int n = 16 * j + ln;
      Frag16 b;
      const _Float16* bp = wT + (size_t)n * K + 32 * s + hi * 8;
      b.q[0] = *(const f32x4*)(bp);
      b.q[1] = *(const f32x4*)(bp + 16);
      acc[j] = WMMA_F16F32(a.h, b.h, acc[j]);
    }
  }

#pragma unroll
  for (int j = 0; j < 2; ++j) {
    int n = 16 * j + ln;
    float sc = scale[n], bi = bias[n];
#pragma unroll
    for (int v = 0; v < 8; ++v) {
      int m = v + hi * 8;
      float val = fmaxf(acc[j][v] * sc + bi, 0.0f);
      out[(gRowBase + rowBase + m) * (size_t)outStride + outOff + n] = (_Float16)val;
    }
  }
}

// ---------------------------------------------------------------------------
// Bilinear (half-pixel) upsample of a4h(128x128x32) and a32h(16x16x32) into
// fusedIn channels [32..63] and [64..95].
// ---------------------------------------------------------------------------
__device__ __forceinline__
void bilin32(const _Float16* __restrict__ src, int inHW, int y, int x,
             _Float16* __restrict__ dst) {
  float s = (float)inHW / (float)FH;
  float sy = (y + 0.5f) * s - 0.5f;
  float sx = (x + 0.5f) * s - 0.5f;
  float fy0 = floorf(sy), fx0 = floorf(sx);
  float wy = sy - fy0, wx = sx - fx0;
  int y0 = min(max((int)fy0, 0), inHW - 1);
  int y1 = min(max((int)fy0 + 1, 0), inHW - 1);
  int x0 = min(max((int)fx0, 0), inHW - 1);
  int x1 = min(max((int)fx0 + 1, 0), inHW - 1);
  const _Float16* p00 = src + ((size_t)y0 * inHW + x0) * 32;
  const _Float16* p01 = src + ((size_t)y0 * inHW + x1) * 32;
  const _Float16* p10 = src + ((size_t)y1 * inHW + x0) * 32;
  const _Float16* p11 = src + ((size_t)y1 * inHW + x1) * 32;
  float w00 = (1.f - wy) * (1.f - wx), w01 = (1.f - wy) * wx;
  float w10 = wy * (1.f - wx),         w11 = wy * wx;
#pragma unroll 8
  for (int c = 0; c < 32; ++c) {
    float v = w00 * (float)p00[c] + w01 * (float)p01[c] +
              w10 * (float)p10[c] + w11 * (float)p11[c];
    dst[c] = (_Float16)v;
  }
}

__global__ void upsample_concat(const _Float16* __restrict__ a4h,
                                const _Float16* __restrict__ a32h,
                                _Float16* __restrict__ fusedIn) {
  int p = blockIdx.x * blockDim.x + threadIdx.x;
  if (p >= FH * FH) return;
  int y = p >> 8, x = p & 255;
  _Float16* dst = fusedIn + (size_t)p * FC;
  bilin32(a4h, 128, y, x, dst + 32);
  bilin32(a32h, 16, y, x, dst + 64);
}

// ---------------------------------------------------------------------------
// 3x3 SAME conv (96->96) + BN affine + ReLU. 16x8 output tile per block,
// 18x10 halo in LDS, 4 waves x 2 M-tiles (each global B frag feeds 2 WMMAs).
// wfT layout: [9][96][96] f16 = [tap][cout][cin].
// ---------------------------------------------------------------------------
__global__ __launch_bounds__(128)
void fuse3x3_wmma(const _Float16* __restrict__ fusedIn, const _Float16* __restrict__ wfT,
                  const float* __restrict__ sf, const float* __restrict__ bf,
                  _Float16* __restrict__ feat) {
  __shared__ __align__(16) _Float16 shIn[180 * FC];   // 18x10 halo, 96 ch

  const int tid = threadIdx.x;
  const int by = blockIdx.y * 16, bx = blockIdx.x * 8;

  // stage halo (zero pad at borders): 180 pixels * 48 dwords
  for (int i = tid; i < 180 * 48; i += 128) {
    int p = i / 48, d = i - p * 48;
    int ty = p / 10, tx = p - ty * 10;
    int gy = by + ty - 1, gx = bx + tx - 1;
    unsigned val = 0;
    if (gy >= 0 && gy < FH && gx >= 0 && gx < FH)
      val = *(const unsigned*)&fusedIn[((size_t)gy * FH + gx) * FC + d * 2];
    *(unsigned*)&shIn[p * FC + d * 2] = val;
  }
  __syncthreads();

  const int wave = tid >> 5, lane = tid & 31;
  const int ln = lane & 15, hi = lane >> 4;
  const int rowBase = wave * 32;          // 2 M-tiles of 16 rows per wave

  // row -> pixel within 16x8 tile
  const int rA0 = rowBase + ln, rA1 = rowBase + 16 + ln;
  const int ly0 = rA0 >> 3, lx0 = rA0 & 7;
  const int ly1 = rA1 >> 3, lx1 = rA1 & 7;

  v8f acc[2][6] = {};
  for (int t = 0; t < 9; ++t) {
    int dy = t / 3, dx = t - dy * 3;
    const _Float16* pix0 = &shIn[((ly0 + dy) * 10 + (lx0 + dx)) * FC];
    const _Float16* pix1 = &shIn[((ly1 + dy) * 10 + (lx1 + dx)) * FC];
#pragma unroll
    for (int s = 0; s < 3; ++s) {
      Frag16 a0, a1;
      a0.q[0] = *(const f32x4*)(pix0 + 32 * s + hi * 8);
      a0.q[1] = *(const f32x4*)(pix0 + 32 * s + hi * 8 + 16);
      a1.q[0] = *(const f32x4*)(pix1 + 32 * s + hi * 8);
      a1.q[1] = *(const f32x4*)(pix1 + 32 * s + hi * 8 + 16);
#pragma unroll
      for (int j = 0; j < 6; ++j) {
        int n = 16 * j + ln;
        Frag16 b;
        const _Float16* bp = wfT + ((size_t)t * FC + n) * FC + 32 * s + hi * 8;
        b.q[0] = *(const f32x4*)(bp);
        b.q[1] = *(const f32x4*)(bp + 16);
        acc[0][j] = WMMA_F16F32(a0.h, b.h, acc[0][j]);
        acc[1][j] = WMMA_F16F32(a1.h, b.h, acc[1][j]);
      }
    }
  }

#pragma unroll
  for (int tt = 0; tt < 2; ++tt) {
#pragma unroll
    for (int j = 0; j < 6; ++j) {
      int n = 16 * j + ln;
      float sc = sf[n], bi = bf[n];
#pragma unroll
      for (int v = 0; v < 8; ++v) {
        int r = rowBase + tt * 16 + v + hi * 8;
        int gy = by + (r >> 3), gx = bx + (r & 7);
        float val = fmaxf(acc[tt][j][v] * sc + bi, 0.0f);
        feat[((size_t)gy * FH + gx) * FC + n] = (_Float16)val;
      }
    }
  }
}

// ---------------------------------------------------------------------------
// LIIF MLP: rows = 4 branches * 512*512 queries. 256 rows per block (8 waves x
// 2 M-tiles). x[128] = feat gather + rel coords (zero pad 98..127);
// MLP 128->128->128->1. h-tile aliases each wave's private slice of shX.
// ---------------------------------------------------------------------------
__global__ __launch_bounds__(256)
void mlp_wmma(const _Float16* __restrict__ feat, const float* __restrict__ coords,
              const _Float16* __restrict__ W0T, const _Float16* __restrict__ W1T,
              const _Float16* __restrict__ w2h, const float* __restrict__ b0,
              const float* __restrict__ b1, const float* __restrict__ b2v,
              float* __restrict__ pred) {
  __shared__ __align__(16) _Float16 shX[256 * MLP_U];       // 64 KB (x, then h)
  __shared__ __align__(16) _Float16 shW0[MLP_U * MLP_U];    // 32 KB
  __shared__ __align__(16) _Float16 shW1[MLP_U * MLP_U];    // 32 KB

  const int tid = threadIdx.x;

  // stage weights (f16, [n][k] layout) into LDS (async when available)
  for (int i = tid; i < 2048; i += 256) {
    copy16_g2l(W0T + i * 8, shW0 + i * 8);
    copy16_g2l(W1T + i * 8, shW1 + i * 8);
  }

  // stage x tile: one row per thread
  {
    const int r = tid;
    size_t R = (size_t)blockIdx.x * 256 + r;
    int br = (int)(R >> 18);
    int q  = (int)(R & (QN - 1));
    float cy = coords[2 * q], cx = coords[2 * q + 1];
    float vx = (br & 2) ? 1.0f : -1.0f;
    float vy = (br & 1) ? 1.0f : -1.0f;
    const float rr = 1.0f / 256.0f, eps = 1e-6f;
    float csy = fminf(fmaxf(cy + vx * rr + eps, -1.0f + 1e-6f), 1.0f - 1e-6f);
    float csx = fminf(fmaxf(cx + vy * rr + eps, -1.0f + 1e-6f), 1.0f - 1e-6f);
    int iy = min(max((int)rintf((csy + 1.0f) * 0.5f * 256.0f - 0.5f), 0), 255);
    int ix = min(max((int)rintf((csx + 1.0f) * 0.5f * 256.0f - 0.5f), 0), 255);
    float fcy = ((iy + 0.5f) / 256.0f) * 2.0f - 1.0f;
    float fcx = ((ix + 0.5f) / 256.0f) * 2.0f - 1.0f;
    float rel0 = (cy - fcy) * 256.0f;
    float rel1 = (cx - fcx) * 256.0f;
    const _Float16* src = feat + ((size_t)iy * FH + ix) * FC;
    _Float16* dst = &shX[r * MLP_U];
#pragma unroll
    for (int i = 0; i < 12; ++i)                    // ch 0..95, 12 x 16B
      copy16_g2l(src + i * 8, dst + i * 8);
    dst[96] = (_Float16)rel0;
    dst[97] = (_Float16)rel1;
    for (int c = 98; c < 128; c += 2)
      *(unsigned*)&dst[c] = 0;                      // zero pad
  }
  WAIT_ASYNC();
  __syncthreads();

  const int wave = tid >> 5, lane = tid & 31;
  const int ln = lane & 15, hi = lane >> 4;
  const int rowBase = wave * 32;                    // 2 M-tiles per wave
  _Float16* shHw = &shX[rowBase * MLP_U];           // wave-private h tile alias

  // ---- GEMM1: x[32x128] @ W0[128x128] ----
  v8f acc[2][8] = {};
#pragma unroll
  for (int s = 0; s < 4; ++s) {
    Frag16 a0, a1;
    const _Float16* ap0 = &shX[(rowBase + ln) * MLP_U + 32 * s + hi * 8];
    const _Float16* ap1 = ap0 + 16 * MLP_U;
    a0.q[0] = *(const f32x4*)(ap0); a0.q[1] = *(const f32x4*)(ap0 + 16);
    a1.q[0] = *(const f32x4*)(ap1); a1.q[1] = *(const f32x4*)(ap1 + 16);
#pragma unroll
    for (int j = 0; j < 8; ++j) {
      int n = 16 * j + ln;
      Frag16 b;
      const _Float16* bp = &shW0[n * MLP_U + 32 * s + hi * 8];
      b.q[0] = *(const f32x4*)(bp);
      b.q[1] = *(const f32x4*)(bp + 16);
      acc[0][j] = WMMA_F16F32(a0.h, b.h, acc[0][j]);
      acc[1][j] = WMMA_F16F32(a1.h, b.h, acc[1][j]);
    }
  }
  // bias + relu -> wave-private h tile (same-wave LDS RAW: DS ops in-order,
  // all A-fragment reads of this region precede these writes in program order)
#pragma unroll
  for (int tt = 0; tt < 2; ++tt) {
#pragma unroll
    for (int j = 0; j < 8; ++j) {
      int n = 16 * j + ln;
      float bn = b0[n];
#pragma unroll
      for (int v = 0; v < 8; ++v) {
        int m = tt * 16 + v + hi * 8;
        shHw[m * MLP_U + n] = (_Float16)fmaxf(acc[tt][j][v] + bn, 0.0f);
      }
    }
  }

  // ---- GEMM2: h0[32x128] @ W1[128x128] ----
  v8f acc2[2][8] = {};
#pragma unroll
  for (int s = 0; s < 4; ++s) {
    Frag16 a0, a1;
    const _Float16* ap0 = &shHw[ln * MLP_U + 32 * s + hi * 8];
    const _Float16* ap1 = ap0 + 16 * MLP_U;
    a0.q[0] = *(const f32x4*)(ap0); a0.q[1] = *(const f32x4*)(ap0 + 16);
    a1.q[0] = *(const f32x4*)(ap1); a1.q[1] = *(const f32x4*)(ap1 + 16);
#pragma unroll
    for (int j = 0; j < 8; ++j) {
      int n = 16 * j + ln;
      Frag16 b;
      const _Float16* bp = &shW1[n * MLP_U + 32 * s + hi * 8];
      b.q[0] = *(const f32x4*)(bp);
      b.q[1] = *(const f32x4*)(bp + 16);
      acc2[0][j] = WMMA_F16F32(a0.h, b.h, acc2[0][j]);
      acc2[1][j] = WMMA_F16F32(a1.h, b.h, acc2[1][j]);
    }
  }

  // ---- layer3: dot(relu(h1 + b1), w2) via VALU + cross-lane reduce ----
  float part[2][8] = {};
#pragma unroll
  for (int j = 0; j < 8; ++j) {
    int n = 16 * j + ln;
    float bn = b1[n];
    float wn = (float)w2h[n];
#pragma unroll
    for (int tt = 0; tt < 2; ++tt)
#pragma unroll
      for (int v = 0; v < 8; ++v)
        part[tt][v] += fmaxf(acc2[tt][j][v] + bn, 0.0f) * wn;
  }
#pragma unroll
  for (int m = 1; m <= 8; m <<= 1) {
#pragma unroll
    for (int tt = 0; tt < 2; ++tt)
#pragma unroll
      for (int v = 0; v < 8; ++v)
        part[tt][v] += __shfl_xor(part[tt][v], m, 32);
  }
  if (ln == 0) {
    float b2s = b2v[0];
#pragma unroll
    for (int tt = 0; tt < 2; ++tt)
#pragma unroll
      for (int v = 0; v < 8; ++v)
        pred[(size_t)blockIdx.x * 256 + rowBase + tt * 16 + v + hi * 8] =
            part[tt][v] + b2s;
  }
}

// ---------------------------------------------------------------------------
// Local ensemble combine (areas recomputed analytically, diagonal swap).
// ---------------------------------------------------------------------------
__global__ void combine_kernel(const float* __restrict__ coords,
                               const float* __restrict__ pred,
                               float* __restrict__ out) {
  int q = blockIdx.x * blockDim.x + threadIdx.x;
  if (q >= QN) return;
  float cy = coords[2 * q], cx = coords[2 * q + 1];
  const float rr = 1.0f / 256.0f, eps = 1e-6f;
  float area[4];
#pragma unroll
  for (int br = 0; br < 4; ++br) {
    float vx = (br & 2) ? 1.0f : -1.0f;
    float vy = (br & 1) ? 1.0f : -1.0f;
    float csy = fminf(fmaxf(cy + vx * rr + eps, -1.0f + 1e-6f), 1.0f - 1e-6f);
    float csx = fminf(fmaxf(cx + vy * rr + eps, -1.0f + 1e-6f), 1.0f - 1e-6f);
    int iy = min(max((int)rintf((csy + 1.0f) * 0.5f * 256.0f - 0.5f), 0), 255);
    int ix = min(max((int)rintf((csx + 1.0f) * 0.5f * 256.0f - 0.5f), 0), 255);
    float fcy = ((iy + 0.5f) / 256.0f) * 2.0f - 1.0f;
    float fcx = ((ix + 0.5f) / 256.0f) * 2.0f - 1.0f;
    float rel0 = (cy - fcy) * 256.0f;
    float rel1 = (cx - fcx) * 256.0f;
    area[br] = fabsf(rel0 * rel1) + 1e-9f;
  }
  float total = area[0] + area[1] + area[2] + area[3];
  float num = pred[0 * QN + q] * area[3] + pred[1 * QN + q] * area[2] +
              pred[2 * QN + q] * area[1] + pred[3 * QN + q] * area[0];
  out[q] = num / total;
}

// ---------------------------------------------------------------------------
// Workspace layout (bytes, 256-aligned)
// ---------------------------------------------------------------------------
#define OFF_FUSEDIN 0u                       // 65536*96 f16 = 12,582,912
#define OFF_FEAT    12582912u                // 65536*96 f16
#define OFF_A4H     25165824u                // 16384*32 f16
#define OFF_A32H    26214400u                // 256*32 f16
#define OFF_W2T     26230784u                // 32*64 f16
#define OFF_W4T     26234880u                // 32*128 f16
#define OFF_W32T    26243072u                // 32*256 f16
#define OFF_WFT     26259456u                // 9*96*96 f16
#define OFF_W0T     26425344u                // 128*128 f16
#define OFF_W1T     26458112u                // 128*128 f16
#define OFF_W2H     26490880u                // 128 f16
#define OFF_PRED    26491136u                // 4*262144 f32

extern "C" void kernel_launch(void* const* d_in, const int* in_sizes, int n_in,
                              void* d_out, int out_size, void* d_ws, size_t ws_size,
                              hipStream_t stream) {
  (void)in_sizes; (void)n_in; (void)out_size; (void)ws_size;

  const float* feats2  = (const float*)d_in[0];
  const float* feats4  = (const float*)d_in[1];
  const float* feats32 = (const float*)d_in[2];
  const float* coords  = (const float*)d_in[3];
  const float* w2  = (const float*)d_in[4];
  const float* s2  = (const float*)d_in[5];
  const float* b2  = (const float*)d_in[6];
  const float* w4  = (const float*)d_in[7];
  const float* s4  = (const float*)d_in[8];
  const float* b4  = (const float*)d_in[9];
  const float* w32 = (const float*)d_in[10];
  const float* s32 = (const float*)d_in[11];
  const float* b32 = (const float*)d_in[12];
  const float* wf  = (const float*)d_in[13];
  const float* sf  = (const float*)d_in[14];
  const float* bf  = (const float*)d_in[15];
  const float* mw0 = (const float*)d_in[16];
  const float* mb0 = (const float*)d_in[17];
  const float* mw1 = (const float*)d_in[18];
  const float* mb1 = (const float*)d_in[19];
  const float* mw2 = (const float*)d_in[20];
  const float* mb2 = (const float*)d_in[21];

  char* ws = (char*)d_ws;
  _Float16* fusedIn = (_Float16*)(ws + OFF_FUSEDIN);
  _Float16* featB   = (_Float16*)(ws + OFF_FEAT);
  _Float16* a4h     = (_Float16*)(ws + OFF_A4H);
  _Float16* a32h    = (_Float16*)(ws + OFF_A32H);
  _Float16* w2T     = (_Float16*)(ws + OFF_W2T);
  _Float16* w4T     = (_Float16*)(ws + OFF_W4T);
  _Float16* w32T    = (_Float16*)(ws + OFF_W32T);
  _Float16* wfT     = (_Float16*)(ws + OFF_WFT);
  _Float16* W0T     = (_Float16*)(ws + OFF_W0T);
  _Float16* W1T     = (_Float16*)(ws + OFF_W1T);
  _Float16* w2h     = (_Float16*)(ws + OFF_W2H);
  float*    pred    = (float*)(ws + OFF_PRED);

  // weight conversion/transpose to f16 [n][kpad]
  cvtT_kernel<<<(32 * 64 + 255) / 256, 256, 0, stream>>>(w2, w2T, 64, 32, 64, 1);
  cvtT_kernel<<<(32 * 128 + 255) / 256, 256, 0, stream>>>(w4, w4T, 128, 32, 128, 1);
  cvtT_kernel<<<(32 * 256 + 255) / 256, 256, 0, stream>>>(w32, w32T, 256, 32, 256, 1);
  cvtT_kernel<<<(9 * 96 * 96 + 255) / 256, 256, 0, stream>>>(wf, wfT, 96, 96, 96, 9);
  cvtT_kernel<<<(128 * 128 + 255) / 256, 256, 0, stream>>>(mw0, W0T, 98, 128, 128, 1);
  cvtT_kernel<<<(128 * 128 + 255) / 256, 256, 0, stream>>>(mw1, W1T, 128, 128, 128, 1);
  cvtT_kernel<<<1, 256, 0, stream>>>(mw2, w2h, 128, 1, 128, 1);

  // ASPP 1x1 convs (WMMA); a2 written directly into fusedIn ch[0..31]
  conv1x1_wmma<<<65536 / 64, 128, 0, stream>>>(feats2, w2T, s2, b2, fusedIn, 64, FC, 0);
  conv1x1_wmma<<<16384 / 64, 128, 0, stream>>>(feats4, w4T, s4, b4, a4h, 128, 32, 0);
  conv1x1_wmma<<<256 / 64, 128, 0, stream>>>(feats32, w32T, s32, b32, a32h, 256, 32, 0);

  // bilinear upsample into fusedIn ch[32..95]
  upsample_concat<<<(FH * FH + 255) / 256, 256, 0, stream>>>(a4h, a32h, fusedIn);

  // 3x3 fuse conv (WMMA), 16x8 tiles
  fuse3x3_wmma<<<dim3(32, 16, 1), 128, 0, stream>>>(fusedIn, wfT, sf, bf, featB);

  // LIIF MLP over 4*512*512 rows (WMMA), 256 rows/block
  mlp_wmma<<<(4 * QN) / 256, 256, 0, stream>>>(featB, coords, W0T, W1T, w2h,
                                               mb0, mb1, mb2, pred);

  // local ensemble combine
  combine_kernel<<<(QN + 255) / 256, 256, 0, stream>>>(coords, pred, (float*)d_out);
}
